// Decoder_48378511622552
// MI455X (gfx1250) — matrix operand order repaired and checked by reading, not compiled, not run on hardware
//
#include <hip/hip_runtime.h>
#include <hip/hip_bf16.h>

#define H 128
#define N_C 50000
#define N_F 150000
#define E_C 400000

typedef __attribute__((ext_vector_type(16))) _Float16 v16h;
typedef __attribute__((ext_vector_type(8)))  float    v8f;

union AFrag { v16h h; unsigned u[8]; };

// branchless ELU: v_exp_f32 + v_cndmask, no EXEC-mask branches
__device__ __forceinline__ float eluf(float v) {
  float e = __expf(fminf(v, 0.f)) - 1.f;
  return v > 0.f ? v : e;
}

// 16x32 f16 A fragment from LDS, row-major with row stride ldk (halfs), K offset k0.
// ISA layout: lanes 0-15 rows, VGPR v<4 -> K = 2v + 8*laneHalf, v>=4 -> K = 16 + 2(v-4) + 8*laneHalf
__device__ __forceinline__ v16h load_a_lds(const _Float16* s, int ldk, int k0) {
  int lane = threadIdx.x & 31;
  int row = lane & 15, half = lane >> 4;
  const _Float16* p = s + row * ldk + k0 + half * 8;
  AFrag f;
#pragma unroll
  for (int v = 0; v < 8; ++v) {
    int kb = ((v & 3) * 2) + ((v >> 2) * 16);
    f.u[v] = *(const unsigned*)(p + kb);
  }
  return f.h;
}

// 32x16 f16 B fragment from transposed weights wT[col*K + k] (col-major of original KxN).
// ISA layout: lane = column, VGPR v<4 -> K = 2v + 8*laneHalf, v>=4 -> K = 16 + 2(v-4) + 8*laneHalf
__device__ __forceinline__ v16h load_b_wt(const _Float16* __restrict__ wT, int K, int k0, int col0) {
  int lane = threadIdx.x & 31;
  int col = col0 + (lane & 15);
  int koff = (lane >> 4) * 8;
  const _Float16* p = wT + col * K + k0 + koff;
  AFrag f;
#pragma unroll
  for (int v = 0; v < 8; ++v) f.u[v] = *(const unsigned*)(p + ((v & 3) * 2) + ((v >> 2) * 16));
  return f.h;
}

__device__ __forceinline__ v8f wmma_f16(v16h a, v16h b, v8f c) {
  return __builtin_amdgcn_wmma_f32_16x16x32_f16(false, a, false, b, (short)0, c,
                                                false, false);
}

// Fused two-stage MLP + residual LayerNorm. 32 rows/block, 8 waves; each wave
// owns one 16-col tile and two 16-row tiles sharing each B fragment.
// MODE 0: edge  A = [x[src[r]] | x[dst[r]] | aux[r]]       (K1=384), res = aux
// MODE 1: node  A = [x[r]      | aux[r]/max(cnt[r],1)]     (K1=256), res = x
// MODE 2: upd   A = [aux[r]    | x[idx0[r]]]               (K1=256), res = aux
template <int K1, int MODE>
__global__ __launch_bounds__(256) void mlp_ln_kernel(
    const float* __restrict__ x, const float* __restrict__ aux,
    const float* __restrict__ cnt, const int* __restrict__ idx0,
    const int* __restrict__ idx1, const _Float16* __restrict__ W0t,
    const float* __restrict__ b0, const _Float16* __restrict__ W1t,
    const float* __restrict__ b1, const float* __restrict__ res,
    const float* __restrict__ g, const float* __restrict__ bt,
    float* __restrict__ out, int nrows) {
  __shared__ _Float16 sA[32 * K1];
  __shared__ _Float16 sH[32 * H];
  __shared__ float    sT[32 * H];
  const int rb = blockIdx.x * 32;
  const int tid = threadIdx.x;
  // cooperative gather + f32->f16 convert into LDS (tail rows clamped)
  for (int i = tid; i < 32 * K1; i += 256) {
    int r = i / K1, c = i % K1;
    int gr = min(rb + r, nrows - 1);
    float v;
    if (MODE == 0) {
      if (c < H)            v = x[idx0[gr] * H + c];
      else if (c < 2 * H)   v = x[idx1[gr] * H + (c - H)];
      else                  v = aux[gr * H + (c - 2 * H)];
    } else if (MODE == 1) {
      if (c < H)            v = x[gr * H + c];
      else                  v = aux[gr * H + (c - H)] / fmaxf(cnt[gr], 1.f);
    } else {
      if (c < H)            v = aux[gr * H + c];
      else                  v = x[idx0[gr] * H + (c - H)];
    }
    sA[i] = (_Float16)v;
  }
  __syncthreads();
  const int wave = tid >> 5;
  const int lane = tid & 31;
  const int col0 = wave * 16;
  const int nl   = lane & 15;
  const int mb   = (lane >> 4) * 8;
  // stage 1: [32 x K1] @ [K1 x 128]; B fragment shared by both row tiles
  v8f acc0 = {}, acc1 = {};
  for (int k0 = 0; k0 < K1; k0 += 32) {
    v16h b  = load_b_wt(W0t, K1, k0, col0);
    v16h a0 = load_a_lds(sA, K1, k0);
    v16h a1 = load_a_lds(sA + 16 * K1, K1, k0);
    acc0 = wmma_f16(a0, b, acc0);
    acc1 = wmma_f16(a1, b, acc1);
  }
  float bias0 = b0[col0 + nl];
#pragma unroll
  for (int v = 0; v < 8; ++v) {
    sH[(v + mb) * H + col0 + nl]      = (_Float16)eluf(acc0[v] + bias0);
    sH[(16 + v + mb) * H + col0 + nl] = (_Float16)eluf(acc1[v] + bias0);
  }
  __syncthreads();
  // stage 2: [32 x 128] @ [128 x 128]
  acc0 = {}; acc1 = {};
  for (int k0 = 0; k0 < H; k0 += 32) {
    v16h b  = load_b_wt(W1t, H, k0, col0);
    v16h a0 = load_a_lds(sH, H, k0);
    v16h a1 = load_a_lds(sH + 16 * H, H, k0);
    acc0 = wmma_f16(a0, b, acc0);
    acc1 = wmma_f16(a1, b, acc1);
  }
  float bias1 = b1[col0 + nl];
#pragma unroll
  for (int v = 0; v < 8; ++v) {
    sT[(v + mb) * H + col0 + nl]      = acc0[v] + bias1;
    sT[(16 + v + mb) * H + col0 + nl] = acc1[v] + bias1;
  }
  __syncthreads();
  // residual + LayerNorm: each wave handles 4 rows, lane covers 4 columns
  for (int rr = 0; rr < 4; ++rr) {
    int r = wave * 4 + rr;
    int gr = rb + r;
    if (gr >= nrows) continue;
    float vals[4], s = 0.f, ss = 0.f;
#pragma unroll
    for (int j = 0; j < 4; ++j) {
      int c = lane + j * 32;
      float v = res[gr * H + c] + sT[r * H + c];
      vals[j] = v; s += v; ss += v * v;
    }
#pragma unroll
    for (int o = 16; o > 0; o >>= 1) {
      s += __shfl_xor(s, o, 32);
      ss += __shfl_xor(ss, o, 32);
    }
    float mu = s * (1.f / H);
    float var = ss * (1.f / H) - mu * mu;
    float rs = rsqrtf(var + 1e-5f);
#pragma unroll
    for (int j = 0; j < 4; ++j) {
      int c = lane + j * 32;
      out[gr * H + c] = (vals[j] - mu) * rs * g[c] + bt[c];
    }
  }
}

// Single GEMM (K=128) + bias + ELU, plain contiguous rows, 32 rows/block.
__global__ __launch_bounds__(256) void gemm_elu_kernel(
    const float* __restrict__ A, const _Float16* __restrict__ Wt,
    const float* __restrict__ b, float* __restrict__ out, int nrows) {
  __shared__ _Float16 sA[32 * H];
  int rb = blockIdx.x * 32, tid = threadIdx.x;
  for (int i = tid; i < 32 * H; i += 256) {
    int gr = min(rb + i / H, nrows - 1);
    sA[i] = (_Float16)A[gr * H + (i & (H - 1))];
  }
  __syncthreads();
  int wave = tid >> 5, lane = tid & 31, col0 = wave * 16;
  int nl = lane & 15, mb = (lane >> 4) * 8;
  v8f acc0 = {}, acc1 = {};
  for (int k0 = 0; k0 < H; k0 += 32) {
    v16h bb = load_b_wt(Wt, H, k0, col0);
    v16h a0 = load_a_lds(sA, H, k0);
    v16h a1 = load_a_lds(sA + 16 * H, H, k0);
    acc0 = wmma_f16(a0, bb, acc0);
    acc1 = wmma_f16(a1, bb, acc1);
  }
  float bias = b[col0 + nl];
#pragma unroll
  for (int v = 0; v < 8; ++v) {
    int r0 = rb + v + mb, r1 = r0 + 16;
    if (r0 < nrows) out[r0 * H + col0 + nl] = eluf(acc0[v] + bias);
    if (r1 < nrows) out[r1 * H + col0 + nl] = eluf(acc1[v] + bias);
  }
}

// convert KxN f32 row-major weight to col-major f16 (wT[n*K+k])
__global__ void cvt_w_kernel(const float* __restrict__ W,
                             _Float16* __restrict__ Wt, int K, int N) {
  int i = blockIdx.x * 256 + threadIdx.x;
  if (i < K * N) {
    int k = i / N, n = i % N;
    Wt[n * K + k] = (_Float16)W[i];
  }
}

__global__ void zero_kernel(float* p, int n) {
  int i = blockIdx.x * 256 + threadIdx.x;
  if (i < n) p[i] = 0.f;
}

// segment-sum scatter: 8 edges per block, 1 wave per edge, lane covers 4 cols
__global__ void scatter_kernel(const float* __restrict__ ea,
                               const int* __restrict__ dst,
                               float* __restrict__ agg, float* __restrict__ cnt) {
  int e = blockIdx.x * 8 + (threadIdx.x >> 5);
  if (e >= E_C) return;
  int lane = threadIdx.x & 31;
  int d = dst[e];
#pragma unroll
  for (int j = 0; j < 4; ++j) {
    int c = lane + j * 32;
    atomicAdd(&agg[d * H + c], ea[e * H + c]);
  }
  if (lane == 0) atomicAdd(&cnt[d], 1.f);
}

// eac_h = elu( (pos_c[cl]-pos_f) @ dW0 + db0 ), K=2 done in VALU
__global__ void pos_h_kernel(const float* __restrict__ pos_c,
                             const float* __restrict__ pos_f,
                             const int* __restrict__ cl,
                             const float* __restrict__ dW0,
                             const float* __restrict__ db0,
                             float* __restrict__ h) {
  int i = blockIdx.x * 256 + threadIdx.x;
  if (i >= N_F * H) return;
  int n = i >> 7, c = i & 127;
  int k = cl[n];
  float p0 = pos_c[k * 2] - pos_f[n * 2];
  float p1 = pos_c[k * 2 + 1] - pos_f[n * 2 + 1];
  h[i] = eluf(p0 * dW0[c] + p1 * dW0[H + c] + db0[c]);
}

// out = elu(x @ oW + ob), N=3
__global__ void out_head_kernel(const float* __restrict__ x,
                                const float* __restrict__ oW,
                                const float* __restrict__ ob,
                                float* __restrict__ out) {
  int i = blockIdx.x * 256 + threadIdx.x;
  if (i >= N_F * 3) return;
  int n = i / 3, j = i % 3;
  float a = ob[j];
#pragma unroll 4
  for (int c = 0; c < H; ++c) a += x[n * H + c] * oW[c * 3 + j];
  out[i] = eluf(a);
}

extern "C" void kernel_launch(void* const* d_in, const int* in_sizes, int n_in,
                              void* d_out, int out_size, void* d_ws,
                              size_t ws_size, hipStream_t stream) {
  (void)in_sizes; (void)n_in; (void)out_size; (void)ws_size;
  const float* x_in   = (const float*)d_in[0];
  const float* ea_in  = (const float*)d_in[1];
  const float* pos_c  = (const float*)d_in[2];
  const float* pos_f  = (const float*)d_in[3];
  const int*   ei     = (const int*)d_in[4];
  const int*   clus   = (const int*)d_in[5];
  const float* eW0 = (const float*)d_in[7];
  const float* eb0 = (const float*)d_in[8];
  const float* eW1 = (const float*)d_in[9];
  const float* eb1 = (const float*)d_in[10];
  const float* eg  = (const float*)d_in[11];
  const float* ebt = (const float*)d_in[12];
  const float* nW0 = (const float*)d_in[13];
  const float* nb0 = (const float*)d_in[14];
  const float* nW1 = (const float*)d_in[15];
  const float* nb1 = (const float*)d_in[16];
  const float* ng  = (const float*)d_in[17];
  const float* nbt = (const float*)d_in[18];
  const float* dW0 = (const float*)d_in[19];
  const float* db0 = (const float*)d_in[20];
  const float* dW1 = (const float*)d_in[21];
  const float* db1 = (const float*)d_in[22];
  const float* uW0 = (const float*)d_in[23];
  const float* ub0 = (const float*)d_in[24];
  const float* uW1 = (const float*)d_in[25];
  const float* ub1 = (const float*)d_in[26];
  const float* ug  = (const float*)d_in[27];
  const float* ubt = (const float*)d_in[28];
  const float* oW  = (const float*)d_in[29];
  const float* ob  = (const float*)d_in[30];
  const int* src = ei;
  const int* dst = ei + E_C;

  // ---- workspace carve-up ----
  char* ws = (char*)d_ws;
  size_t off = 0;
  auto carve = [&](size_t bytes) -> void* {
    void* p = ws + off;
    off += (bytes + 255) & ~(size_t)255;
    return p;
  };
  _Float16* eW0t = (_Float16*)carve((size_t)2 * 384 * H * 2);
  _Float16* eW1t = (_Float16*)carve((size_t)2 * H * H * 2);
  _Float16* nW0t = (_Float16*)carve((size_t)2 * 256 * H * 2);
  _Float16* nW1t = (_Float16*)carve((size_t)2 * H * H * 2);
  _Float16* uW0t = (_Float16*)carve((size_t)256 * H * 2);
  _Float16* uW1t = (_Float16*)carve((size_t)H * H * 2);
  _Float16* dW1t = (_Float16*)carve((size_t)H * H * 2);
  float* xA  = (float*)carve((size_t)N_F * H * 4);
  float* xB  = (float*)carve((size_t)N_F * H * 4);
  float* eaA = (float*)carve((size_t)E_C * H * 4);
  float* eaB = (float*)carve((size_t)E_C * H * 4);
  float* agg = (float*)carve((size_t)N_F * H * 4);
  float* cnt = (float*)carve((size_t)N_F * 4);

  auto cvt = [&](const float* W, _Float16* Wt, int K, int N) {
    int n = K * N;
    cvt_w_kernel<<<(n + 255) / 256, 256, 0, stream>>>(W, Wt, K, N);
  };
  for (int i = 0; i < 2; ++i) {
    cvt(eW0 + (size_t)i * 384 * H, eW0t + (size_t)i * 384 * H, 384, H);
    cvt(eW1 + (size_t)i * H * H,   eW1t + (size_t)i * H * H,   H,   H);
    cvt(nW0 + (size_t)i * 256 * H, nW0t + (size_t)i * 256 * H, 256, H);
    cvt(nW1 + (size_t)i * H * H,   nW1t + (size_t)i * H * H,   H,   H);
  }
  cvt(uW0, uW0t, 256, H);
  cvt(uW1, uW1t, H, H);
  cvt(dW1, dW1t, H, H);

  hipMemcpyAsync(xA, x_in, (size_t)N_C * H * 4, hipMemcpyDeviceToDevice, stream);
  hipMemcpyAsync(eaA, ea_in, (size_t)E_C * H * 4, hipMemcpyDeviceToDevice, stream);

  float *xcur = xA, *xnxt = xB, *eacur = eaA, *eanxt = eaB;
  auto mp_pass = [&](int nnodes) {
    for (int i = 0; i < 2; ++i) {
      mlp_ln_kernel<384, 0><<<(E_C + 31) / 32, 256, 0, stream>>>(
          xcur, eacur, nullptr, src, dst, eW0t + (size_t)i * 384 * H,
          eb0 + i * H, eW1t + (size_t)i * H * H, eb1 + i * H, eacur, eg + i * H,
          ebt + i * H, eanxt, E_C);
      zero_kernel<<<(nnodes * H + 255) / 256, 256, 0, stream>>>(agg, nnodes * H);
      zero_kernel<<<(nnodes + 255) / 256, 256, 0, stream>>>(cnt, nnodes);
      scatter_kernel<<<E_C / 8, 256, 0, stream>>>(eanxt, dst, agg, cnt);
      mlp_ln_kernel<256, 1><<<(nnodes + 31) / 32, 256, 0, stream>>>(
          xcur, agg, cnt, nullptr, nullptr, nW0t + (size_t)i * 256 * H,
          nb0 + i * H, nW1t + (size_t)i * H * H, nb1 + i * H, xcur, ng + i * H,
          nbt + i * H, xnxt, nnodes);
      float* t;
      t = xcur; xcur = xnxt; xnxt = t;
      t = eacur; eacur = eanxt; eanxt = t;
    }
  };

  // pass 1 on coarse graph (x: N_C rows)
  mp_pass(N_C);
  // after pass 1: xcur == xA (coarse x), eacur == eaA; xB/eaB free

  // decoder: pos MLP -> eac (reuse free eaB region), then update MLP -> xB
  float* eac = eanxt;  // == eaB, free between passes, N_F*H fits in E_C*H
  pos_h_kernel<<<(N_F * H + 255) / 256, 256, 0, stream>>>(pos_c, pos_f, clus,
                                                          dW0, db0, agg);
  gemm_elu_kernel<<<(N_F + 31) / 32, 256, 0, stream>>>(agg, dW1t, db1, eac, N_F);
  // t = LN(eac + MLP([eac | x_c[clusters]])); f2c seg_mean is identity
  mlp_ln_kernel<256, 2><<<(N_F + 31) / 32, 256, 0, stream>>>(
      xcur, eac, nullptr, clus, nullptr, uW0t, ub0, uW1t, ub1, eac, ug, ubt,
      xnxt, N_F);
  // x for pass 2 lives in xnxt (== xB): make it current
  { float* t = xcur; xcur = xnxt; xnxt = t; }

  // pass 2 on fine nodes (x: N_F rows, same edges)
  mp_pass(N_F);
  // final x in xcur

  out_head_kernel<<<(N_F * 3 + 255) / 256, 256, 0, stream>>>(xcur, oW, ob,
                                                             (float*)d_out);
  // second tuple element: edge_index_c passthrough (raw int bits)
  hipMemcpyAsync((char*)d_out + (size_t)N_F * 3 * 4, (const void*)ei,
                 (size_t)2 * E_C * 4, hipMemcpyDeviceToDevice, stream);
}
